// CrossBlock_53309134078438
// MI455X (gfx1250) — matrix operand order repaired and checked by reading, not compiled, not run on hardware
//
#include <hip/hip_runtime.h>
#include <hip/hip_bf16.h>
#include <math.h>

// ---------------------------------------------------------------------------
// Constants for this problem instance (shapes fixed by the reference module)
// ---------------------------------------------------------------------------
#define BB   2
#define VV   5
#define HH   60
#define WW2  80
#define CC   256
#define NHH  8
#define HWL  4801                 // H*W + 1
#define MROWS (BB*VV*HWL)         // 48010
#define MPAD  48032               // MROWS rounded up to 32 (2x2 wave tiling)
#define LVIEW (VV*HWL)            // 24005 rows per batch for attention
#define LOG10000 9.210340371976184f
#define PI_F 3.14159265358979323846f

typedef __bf16 bf16_t;
typedef __attribute__((ext_vector_type(16))) __bf16 v16bf;
typedef __attribute__((ext_vector_type(8)))  __bf16 v8bf;
typedef __attribute__((ext_vector_type(8)))  float  v8f;

// ---------------------------------------------------------------------------
// Repack fp32 weight W[K][N] (row-major) into bf16 WMMA B-fragments.
// Fragment (nTile16,kTile): lane 0-15  -> col = nTile16*16+lane,    K = kTile*32+{0..7,16..23}
//                           lane 16-31 -> col = nTile16*16+lane-16, K = kTile*32+{8..15,24..31}
// 16 bf16 elems per lane stored contiguously -> GEMM does 2x b128 loads/lane.
// ---------------------------------------------------------------------------
__global__ __launch_bounds__(256) void repack_w(const float* __restrict__ W,
                                                bf16_t* __restrict__ F,
                                                int K, int N) {
  int t = blockIdx.x * 256 + threadIdx.x;
  int total = (N >> 4) * (K >> 5) * 32;
  if (t >= total) return;
  int lane  = t & 31;
  int frag  = t >> 5;
  int kTiles = K >> 5;
  int nTile = frag / kTiles;
  int kTile = frag - nTile * kTiles;
  int col  = nTile * 16 + (lane & 15);
  int koff = kTile * 32 + ((lane < 16) ? 0 : 8);
  bf16_t* dst = F + (size_t)t * 16;
#pragma unroll
  for (int j = 0; j < 16; ++j)
    dst[j] = (bf16_t)W[(size_t)(koff + (j < 8 ? j : j + 8)) * N + col];
}

// ---------------------------------------------------------------------------
// Fused epipolar sine embeddings + residual add: x1 = x + tok_emb  (bf16 out)
// ---------------------------------------------------------------------------
__global__ __launch_bounds__(256) void embed_kernel(const float* __restrict__ x,
                                                    const float* __restrict__ epipole,
                                                    const float* __restrict__ tok_table,
                                                    bf16_t* __restrict__ x1b) {
  int idx = blockIdx.x * 256 + threadIdx.x;
  if (idx >= MROWS * CC) return;
  int c   = idx & (CC - 1);
  int row = idx >> 8;
  int bv  = row / HWL;
  int pos = row - bv * HWL;
  int b    = bv / VV;
  int view = bv - b * VV;
  float tokv;
  if (view == 0) {
    tokv = tok_table[c];                       // ref token, all HW positions
  } else if (pos == 0) {
    tokv = tok_table[CC + c];                  // sep token
  } else {
    int nv = view - 1;
    int p  = pos - 1;
    int py = p / WW2;
    int px = p - py * WW2;
    float ex = epipole[((b * (VV - 1)) + nv) * 2 + 0];
    float ey = epipole[((b * (VV - 1)) + nv) * 2 + 1];
    float val, scale, t;
    int j, odd;
    if (c < 64) {            // dir_emb: _sine2(abs_u, abs_v, 32, 2*pi) -> 64 ch
      float en = fmaxf(sqrtf(ex * ex + ey * ey), 1e-12f);
      int cc = c & 31;
      val = (c < 32) ? (ey / en) : (ex / en);  // ev first, then eu
      j = cc >> 1; odd = cc & 1;
      t = __expf((2.0f * (float)j / 32.0f) * LOG10000);
      scale = 2.0f * PI_F;
    } else if (c < 128) {    // dis_emb: _sine1(dis, 64, 2*pi) -> 64 ch
      float dn = sqrtf(ex * ex + ey * ey) * (1.0f / 512.0f);
      val = fminf(fmaxf(dn, 0.0f), 1.0f);
      int cc = c - 64;
      j = cc >> 1; odd = cc & 1;
      t = __expf((2.0f * (float)j / 64.0f) * LOG10000);
      scale = 2.0f * PI_F;
    } else {                 // rel_emb: _sine2(rel_u, rel_v, 64, 32*pi) -> 128 ch
      float ru = (float)px - ex;
      float rv = (float)py - ey;
      float nr = sqrtf(ru * ru + rv * rv) + 1e-6f;
      int cc = c - 128;
      val = (cc < 64) ? (rv / nr) : (ru / nr);
      int c6 = cc & 63;
      j = c6 >> 1; odd = c6 & 1;
      t = __expf((2.0f * (float)j / 64.0f) * LOG10000);
      scale = 32.0f * PI_F;
    }
    float ph = val * scale / t;
    tokv = odd ? cosf(ph) : sinf(ph);
  }
  x1b[idx] = (bf16_t)(x[idx] + tokv);
}

// ---------------------------------------------------------------------------
// Per-element GEMM epilogues
// EPI 1: qkv  -> elu(q,k)+1, store bf16
// EPI 2: proj -> +bias, x2 = x + a1*out, store fp32 + bf16
// EPI 3: fc1  -> +bias, exact-erf GELU, store bf16
// EPI 4: fc2  -> +bias, out = x2 + a2*mlp, store fp32 (d_out)
// ---------------------------------------------------------------------------
template <int EPI>
__device__ __forceinline__ void epilogue_elem(float v, int grow, int gcol,
                                              int M, int N, float al,
                                              const float* __restrict__ bias,
                                              const float* __restrict__ resid,
                                              float* __restrict__ outF,
                                              bf16_t* __restrict__ outB) {
  if (grow >= M) return;                        // never store pad rows
  size_t o = (size_t)grow * N + gcol;
  if (EPI == 1) {
    if (gcol < 2 * CC) v = (v > 0.0f) ? (v + 1.0f) : __expf(v);  // elu(x)+1
    outB[o] = (bf16_t)v;
  } else if (EPI == 2) {
    v += bias[gcol];
    float o2 = resid[o] + al * v;               // x + alpha1*attn_out
    outF[o] = o2;
    outB[o] = (bf16_t)o2;
  } else if (EPI == 3) {
    v += bias[gcol];
    float g = 0.5f * v * (1.0f + erff(v * 0.70710678118654752f));
    outB[o] = (bf16_t)g;
  } else {
    v += bias[gcol];
    outF[o] = resid[o] + al * v;                // x2 + alpha2*mlp_out
  }
}

// ---------------------------------------------------------------------------
// WMMA bf16 GEMM with 2x2 register tiling: one wave computes a 32x32 fp32
// tile. Per 32-wide k-step: 8x global_load_b128 feed 4 independent
// v_wmma_f32_16x16x32_bf16 -> 2x compute per loaded byte vs 1x1 tiling and
// 4 independent accumulation chains to hide XDL latency.
// ---------------------------------------------------------------------------
template <int EPI>
__global__ __launch_bounds__(256) void gemm_wmma(const bf16_t* __restrict__ A,
                                                 const bf16_t* __restrict__ Bf,
                                                 const float* __restrict__ bias,
                                                 const float* __restrict__ resid,
                                                 const float* __restrict__ alpha,
                                                 float* __restrict__ outF,
                                                 bf16_t* __restrict__ outB,
                                                 int M, int K, int N,
                                                 int tilesM, int tilesN) {
  int lane = threadIdx.x & 31;
  int gw   = blockIdx.x * (blockDim.x >> 5) + (threadIdx.x >> 5);
  if (gw >= tilesM * tilesN) return;
  int mT = gw / tilesN;          // 32-row tile
  int nT = gw - mT * tilesN;     // 32-col tile

  // A fragment addressing (ISA 16-bit A 16x32 layout):
  // lanes 0-15: row = lane,   elems 0..7 -> K k0+0..7,  elems 8..15 -> K k0+16..23
  // lanes16-31: row = lane-16,elems 0..7 -> K k0+8..15, elems 8..15 -> K k0+24..31
  int rowA = mT * 32 + (lane & 15);
  const bf16_t* aRow0 = A + (size_t)rowA * K + ((lane < 16) ? 0 : 8);
  const bf16_t* aRow1 = aRow0 + (size_t)16 * K;
  const int kTiles = K >> 5;
  const bf16_t* bBase0 = Bf + ((size_t)(nT * 2 + 0) * kTiles * 32 + lane) * 16;
  const bf16_t* bBase1 = Bf + ((size_t)(nT * 2 + 1) * kTiles * 32 + lane) * 16;

  v8f acc00 = {}, acc01 = {}, acc10 = {}, acc11 = {};
  for (int k0 = 0; k0 < K; k0 += 32) {
    size_t bOff = (size_t)(k0 >> 5) * (32 * 16);
    v8bf a0lo = *(const v8bf*)(aRow0 + k0);
    v8bf a0hi = *(const v8bf*)(aRow0 + k0 + 16);
    v8bf a1lo = *(const v8bf*)(aRow1 + k0);
    v8bf a1hi = *(const v8bf*)(aRow1 + k0 + 16);
    v8bf b0lo = *(const v8bf*)(bBase0 + bOff);
    v8bf b0hi = *(const v8bf*)(bBase0 + bOff + 8);
    v8bf b1lo = *(const v8bf*)(bBase1 + bOff);
    v8bf b1hi = *(const v8bf*)(bBase1 + bOff + 8);
    v16bf a0, a1, b0, b1;
#pragma unroll
    for (int i = 0; i < 8; ++i) {
      a0[i] = a0lo[i]; a0[i + 8] = a0hi[i];
      a1[i] = a1lo[i]; a1[i + 8] = a1hi[i];
      b0[i] = b0lo[i]; b0[i + 8] = b0hi[i];
      b1[i] = b1lo[i]; b1[i + 8] = b1hi[i];
    }
    acc00 = __builtin_amdgcn_wmma_f32_16x16x32_bf16(false, a0, false, b0,
                                                    (short)0, acc00, false, false);
    acc01 = __builtin_amdgcn_wmma_f32_16x16x32_bf16(false, a0, false, b1,
                                                    (short)0, acc01, false, false);
    acc10 = __builtin_amdgcn_wmma_f32_16x16x32_bf16(false, a1, false, b0,
                                                    (short)0, acc10, false, false);
    acc11 = __builtin_amdgcn_wmma_f32_16x16x32_bf16(false, a1, false, b1,
                                                    (short)0, acc11, false, false);
  }

  // C/D layout: elem r -> row = r (+8 for lanes 16-31), col = lane&15
  int   colL  = lane & 15;
  int   rsub  = (lane < 16) ? 0 : 8;
  float al    = (EPI == 2 || EPI == 4) ? alpha[0] : 0.0f;
#pragma unroll
  for (int r = 0; r < 8; ++r) {
    int row0 = mT * 32 + rsub + r;        // M sub-tile 0
    int row1 = row0 + 16;                 // M sub-tile 1
    int col0 = nT * 32 + colL;            // N sub-tile 0
    int col1 = col0 + 16;                 // N sub-tile 1
    epilogue_elem<EPI>(acc00[r], row0, col0, M, N, al, bias, resid, outF, outB);
    epilogue_elem<EPI>(acc01[r], row0, col1, M, N, al, bias, resid, outF, outB);
    epilogue_elem<EPI>(acc10[r], row1, col0, M, N, al, bias, resid, outF, outB);
    epilogue_elem<EPI>(acc11[r], row1, col1, M, N, al, bias, resid, outF, outB);
  }
}

// ---------------------------------------------------------------------------
// kv[n][h][m][d] = sum_l k'[l,h,d] * v[l,h,m];  ksum[n][h][d] = sum_l k'[l,h,d]
// fp32 register accumulation over row chunks + global fp32 atomics (tiny output)
// ---------------------------------------------------------------------------
__global__ __launch_bounds__(256) void kv_reduce(const bf16_t* __restrict__ qkv,
                                                 float* __restrict__ kv,
                                                 float* __restrict__ ksum) {
  int h = blockIdx.y, n = blockIdx.z;
  int r0 = blockIdx.x * 2048;
  int r1 = r0 + 2048; if (r1 > LVIEW) r1 = LVIEW;
  int t = threadIdx.x;
  int d  = t & 31;
  int mb = (t >> 5) << 2;        // each thread owns 4 consecutive m values
  float a0 = 0.f, a1 = 0.f, a2 = 0.f, a3 = 0.f, ks = 0.f;
  for (int r = r0; r < r1; ++r) {
    size_t base = (size_t)(n * LVIEW + r) * (3 * CC);
    float kval = (float)qkv[base + CC + h * 32 + d];
    const bf16_t* vp = qkv + base + 2 * CC + h * 32 + mb;
    a0 += kval * (float)vp[0];
    a1 += kval * (float)vp[1];
    a2 += kval * (float)vp[2];
    a3 += kval * (float)vp[3];
    if (t < 32) ks += kval;
  }
  float* kvb = kv + (size_t)((n * NHH + h) * 32) * 32;   // [m][d]
  atomicAdd(&kvb[(mb + 0) * 32 + d], a0);
  atomicAdd(&kvb[(mb + 1) * 32 + d], a1);
  atomicAdd(&kvb[(mb + 2) * 32 + d], a2);
  atomicAdd(&kvb[(mb + 3) * 32 + d], a3);
  if (t < 32) atomicAdd(&ksum[(n * NHH + h) * 32 + d], ks);
}

// ---------------------------------------------------------------------------
// y[l,h,m] = z * sum_d q'[l,h,d] * kv[n,h,m,d];  z = 1/(q'.ksum + eps)
// one wave32 per (row, head); lane = m; q broadcast via shuffles
// ---------------------------------------------------------------------------
__global__ __launch_bounds__(256) void attn_y(const bf16_t* __restrict__ qkv,
                                              const float* __restrict__ kv,
                                              const float* __restrict__ ksum,
                                              bf16_t* __restrict__ yb) {
  int lane = threadIdx.x & 31;
  int grp  = blockIdx.x * 8 + (threadIdx.x >> 5);
  if (grp >= MROWS * NHH) return;
  int row = grp >> 3;
  int h   = grp & 7;
  int n   = (row >= LVIEW) ? 1 : 0;
  float q = (float)qkv[(size_t)row * (3 * CC) + h * 32 + lane];
  float p = q * ksum[(n * NHH + h) * 32 + lane];
#pragma unroll
  for (int off = 16; off; off >>= 1) p += __shfl_xor(p, off, 32);
  float z = 1.0f / (p + 1e-6f);
  const float* kvm = kv + (size_t)((n * NHH + h) * 32 + lane) * 32;  // m=lane, over d
  float acc = 0.0f;
#pragma unroll
  for (int dd = 0; dd < 32; ++dd) acc += __shfl(q, dd, 32) * kvm[dd];
  yb[(size_t)row * CC + h * 32 + lane] = (bf16_t)(acc * z);
}

// ---------------------------------------------------------------------------
extern "C" void kernel_launch(void* const* d_in, const int* in_sizes, int n_in,
                              void* d_out, int out_size, void* d_ws, size_t ws_size,
                              hipStream_t stream) {
  const float* x      = (const float*)d_in[0];
  const float* epi    = (const float*)d_in[1];
  const float* w_qkv  = (const float*)d_in[2];
  const float* w_proj = (const float*)d_in[3];
  const float* b_proj = (const float*)d_in[4];
  const float* w_fc1  = (const float*)d_in[5];
  const float* b_fc1  = (const float*)d_in[6];
  const float* w_fc2  = (const float*)d_in[7];
  const float* b_fc2  = (const float*)d_in[8];
  const float* tok    = (const float*)d_in[9];
  const float* alpha1 = (const float*)d_in[10];
  const float* alpha2 = (const float*)d_in[11];
  float* out = (float*)d_out;

  char* ws = (char*)d_ws;
  size_t off = 0;
  auto take = [&](size_t bytes) {
    size_t cur = off;
    off += (bytes + 255) & ~(size_t)255;
    return cur;
  };
  bf16_t* wqkvF  = (bf16_t*)(ws + take((size_t)CC * 3 * CC * 2));
  bf16_t* wprojF = (bf16_t*)(ws + take((size_t)CC * CC * 2));
  bf16_t* wfc1F  = (bf16_t*)(ws + take((size_t)CC * 2 * CC * 2));
  bf16_t* wfc2F  = (bf16_t*)(ws + take((size_t)2 * CC * CC * 2));
  bf16_t* x1b    = (bf16_t*)(ws + take((size_t)MPAD * CC * 2));
  bf16_t* qkvb   = (bf16_t*)(ws + take((size_t)MPAD * 3 * CC * 2));
  float*  kvbuf  = (float*)(ws + take((size_t)BB * NHH * 32 * 32 * 4));
  float*  ksum   = (float*)(ws + take((size_t)BB * NHH * 32 * 4));
  bf16_t* yb     = (bf16_t*)(ws + take((size_t)MPAD * CC * 2));
  float*  x2f    = (float*)(ws + take((size_t)MPAD * CC * 4));
  bf16_t* x2b    = (bf16_t*)(ws + take((size_t)MPAD * CC * 2));
  bf16_t* hb     = (bf16_t*)(ws + take((size_t)MPAD * 2 * CC * 2));

  const int tilesM = MPAD / 32;  // 1501 (32-row wave tiles)

  // Weight repack to WMMA fragment layout (bf16)
  repack_w<<<48, 256, 0, stream>>>(w_qkv,  wqkvF,  CC,     3 * CC);
  repack_w<<<16, 256, 0, stream>>>(w_proj, wprojF, CC,     CC);
  repack_w<<<32, 256, 0, stream>>>(w_fc1,  wfc1F,  CC,     2 * CC);
  repack_w<<<32, 256, 0, stream>>>(w_fc2,  wfc2F,  2 * CC, CC);

  // x1 = x + tok_emb (fused sine embeddings), bf16
  embed_kernel<<<(MROWS * CC + 255) / 256, 256, 0, stream>>>(x, epi, tok, x1b);

  // GEMM1: qkv = x1 @ w_qkv, epilogue elu+1 on q,k
  {
    int tN = (3 * CC) / 32;  // 24
    gemm_wmma<1><<<(tilesM * tN + 7) / 8, 256, 0, stream>>>(
        x1b, wqkvF, b_proj, x, alpha1, x2f, qkvb, MROWS, CC, 3 * CC, tilesM, tN);
  }

  // zero kv/ksum accumulators (atomics each call)
  hipMemsetAsync(kvbuf, 0, (size_t)BB * NHH * 32 * 32 * 4, stream);
  hipMemsetAsync(ksum,  0, (size_t)BB * NHH * 32 * 4, stream);

  // kv = k'^T v, ksum = sum_l k'
  {
    dim3 grid((LVIEW + 2047) / 2048, NHH, BB);
    kv_reduce<<<grid, 256, 0, stream>>>(qkvb, kvbuf, ksum);
  }

  // y = normalize(q' kv)
  attn_y<<<(MROWS * NHH + 7) / 8, 256, 0, stream>>>(qkvb, kvbuf, ksum, yb);

  // GEMM2: x2 = x + a1*(y @ w_proj + b_proj), dual fp32/bf16 store
  {
    int tN = CC / 32;  // 8
    gemm_wmma<2><<<(tilesM * tN + 7) / 8, 256, 0, stream>>>(
        yb, wprojF, b_proj, x, alpha1, x2f, x2b, MROWS, CC, CC, tilesM, tN);
  }

  // GEMM3: h = gelu(x2 @ w_fc1 + b_fc1), bf16
  {
    int tN = (2 * CC) / 32;  // 16
    gemm_wmma<3><<<(tilesM * tN + 7) / 8, 256, 0, stream>>>(
        x2b, wfc1F, b_fc1, x2f, alpha2, x2f, hb, MROWS, CC, 2 * CC, tilesM, tN);
  }

  // GEMM4: out = x2 + a2*(h @ w_fc2 + b_fc2), fp32 -> d_out
  {
    int tN = CC / 32;  // 8
    gemm_wmma<4><<<(tilesM * tN + 7) / 8, 256, 0, stream>>>(
        hb, wfc2F, b_fc2, x2f, alpha2, out, x2b, MROWS, 2 * CC, CC, tilesM, tN);
  }
}
